// GptOSSAttention_28638841929804
// MI455X (gfx1250) — compile-verified, matrix operand order checked by reading, not run on hardware
//
#include <hip/hip_runtime.h>
#include <hip/hip_bf16.h>
#include <math.h>

// ---------------------------------------------------------------------------
// GPT-OSS sliding-window GQA attention layer for MI455X (gfx1250, wave32).
// f16 WMMA (V_WMMA_F32_16X16X32_F16), f32 accumulate.  Weights live in the
// 192MB L2 as f16; GEMM waves compute 32x64 register tiles (8 WMMAs / 12
// b128 loads).  YaRN RoPE + layout split fused into the QKV GEMM epilogue.
// Attention double-buffers K/V tiles into LDS via async-to-LDS copies.
// ---------------------------------------------------------------------------

typedef __attribute__((ext_vector_type(16))) _Float16 v16h;
typedef __attribute__((ext_vector_type(8)))  _Float16 v8h;
typedef __attribute__((ext_vector_type(8)))  float    v8f;
typedef int v4i __attribute__((vector_size(16)));

#define AS1 __attribute__((address_space(1)))
#define AS3 __attribute__((address_space(3)))

#if defined(__has_builtin)
#if __has_builtin(__builtin_amdgcn_global_load_async_to_lds_b128)
#define USE_ASYNC_LDS 1
#endif
#endif

#define DIM       2880
#define N_HEADS   64
#define N_KV      8
#define HEAD_DIM  64
#define WINDOW    128
#define SEQ       1536
#define QKV_DIM   5120      // 64*64 + 2*8*64
#define Q_DIM     4096
#define KV_DIM    512
#define SM_SCALE  0.125f
#define EPS       1e-5f

union FragAB { v16h v; v8h h2[2]; };
union FragC  { v8f  v; float f[8]; };

template <int N>
__device__ __forceinline__ void wait_asynccnt() {
#if __has_builtin(__builtin_amdgcn_s_wait_asynccnt)
  __builtin_amdgcn_s_wait_asynccnt((unsigned short)N);
#else
  asm volatile("s_wait_asynccnt %0" :: "n"(N) : "memory");
#endif
}

__device__ __forceinline__ void wait_dscnt0() {
  asm volatile("s_wait_dscnt 0" ::: "memory");
}

#if defined(USE_ASYNC_LDS)
// 16-byte async global->LDS copy (per active lane).
__device__ __forceinline__ void async_ld16(const _Float16* gsrc, _Float16* ldst) {
  AS1 void* gv = (AS1 void*)(void*)const_cast<_Float16*>(gsrc);
  AS3 void* lv = (AS3 void*)(void*)ldst;
  __builtin_amdgcn_global_load_async_to_lds_b128((AS1 v4i*)gv, (AS3 v4i*)lv, 0, 0);
}
#endif

// ---------------------------------------------------------------------------
// RMSNorm + cast to f16.
// ---------------------------------------------------------------------------
__global__ void __launch_bounds__(256)
rmsnorm_cast(const float* __restrict__ x, const float* __restrict__ w,
             _Float16* __restrict__ t) {
  __shared__ float red[256];
  const int s = blockIdx.x;
  const float* row = x + (size_t)s * DIM;
  float acc = 0.f;
  for (int i = threadIdx.x; i < DIM; i += 256) {
    float v = row[i];
    acc += v * v;
  }
  red[threadIdx.x] = acc;
  __syncthreads();
  for (int o = 128; o > 0; o >>= 1) {
    if ((int)threadIdx.x < o) red[threadIdx.x] += red[threadIdx.x + o];
    __syncthreads();
  }
  const float rs = rsqrtf(red[0] / (float)DIM + EPS);
  for (int i = threadIdx.x; i < DIM; i += 256)
    t[(size_t)s * DIM + i] = (_Float16)(row[i] * rs * w[i]);
}

__global__ void cast_f16(const float* __restrict__ src,
                         _Float16* __restrict__ dst, int n) {
  int i = blockIdx.x * blockDim.x + threadIdx.x;
  if (i < n) dst[i] = (_Float16)src[i];
}

// ---------------------------------------------------------------------------
// YaRN cos/sin for rotation pair index p (0..31) at position pos.
// ---------------------------------------------------------------------------
__device__ __forceinline__ void yarn_cs(int p, float pos, float& c, float& sn) {
  const float freq  = __powf(150000.0f, (float)p * (1.0f / 32.0f));
  const float conc  = 0.1f * __logf(32.0f) + 1.0f;
  const float lb    = __logf(150000.0f);
  const float twopi = 6.2831853071795865f;
  const float low   = 32.0f * __logf(4096.0f / (32.0f * twopi)) / lb; // beta=32
  const float high  = 32.0f * __logf(4096.0f / twopi) / lb;           // alpha=1
  float ramp = ((float)p - low) / (high - low);
  ramp = fminf(fmaxf(ramp, 0.0f), 1.0f);
  const float mask = 1.0f - ramp;
  const float invf = (1.0f / (32.0f * freq)) * (1.0f - mask) +
                     (1.0f / freq) * mask;
  const float ang = pos * invf;
  c  = __cosf(ang) * conc;
  sn = __sinf(ang) * conc;
}

// ---------------------------------------------------------------------------
// GEMM: A[M,K]*W[N,K]^T + bias.  8 waves m-stacked, wave tile 32x64
// (2 A-frags x 4 B-frags -> 8 WMMAs per K-step).  Block tile 256x64.
// ROPE=true: QKV projection; epilogue applies YaRN RoPE and writes
//   q -> [64][S][64], k -> [8][S][64], v -> [8][64][S] (transposed), all f16.
//   (RoPE pair (d,d+32) = accumulator tiles g and g+2, same lane/VGPR.)
// ROPE=false: plain f32 store (output projection -> d_out).
// ---------------------------------------------------------------------------
template <bool ROPE>
__global__ void __launch_bounds__(256)
gemm_nt_f16(const _Float16* __restrict__ A, const _Float16* __restrict__ W,
            const float* __restrict__ bias, float* __restrict__ C,
            _Float16* __restrict__ qf, _Float16* __restrict__ kf,
            _Float16* __restrict__ vT, int M, int N, int K) {
  const int lane = threadIdx.x & 31;
  const int wv   = threadIdx.x >> 5;
  const int m0   = blockIdx.y * 256 + wv * 32;
  const int n0   = blockIdx.x * 64;
  const int l15  = lane & 15;
  const int koff = (lane >> 4) * 8;

  const _Float16* pa0 = A + (size_t)(m0 + l15) * K + koff;
  const _Float16* pa1 = pa0 + (size_t)16 * K;
  const _Float16* pb0 = W + (size_t)(n0 + l15) * K + koff;

  FragC acc[2][4];
#pragma unroll
  for (int mi = 0; mi < 2; ++mi)
#pragma unroll
    for (int g = 0; g < 4; ++g) acc[mi][g].v = (v8f){};

  for (int k0 = 0; k0 < K; k0 += 32) {
    __builtin_prefetch(pa0 + k0 + 512, 0, 3);
    __builtin_prefetch(pa1 + k0 + 512, 0, 3);
    FragAB a0, a1, b[4];
    a0.h2[0] = *(const v8h*)(pa0 + k0);
    a0.h2[1] = *(const v8h*)(pa0 + k0 + 16);
    a1.h2[0] = *(const v8h*)(pa1 + k0);
    a1.h2[1] = *(const v8h*)(pa1 + k0 + 16);
#pragma unroll
    for (int g = 0; g < 4; ++g) {
      const _Float16* pb = pb0 + (size_t)(g * 16) * K + k0;
      __builtin_prefetch(pb + 512, 0, 3);
      b[g].h2[0] = *(const v8h*)pb;
      b[g].h2[1] = *(const v8h*)(pb + 16);
    }
#pragma unroll
    for (int g = 0; g < 4; ++g) {
      acc[0][g].v = __builtin_amdgcn_wmma_f32_16x16x32_f16(
          false, a0.v, false, b[g].v, (short)0, acc[0][g].v, false, false);
      acc[1][g].v = __builtin_amdgcn_wmma_f32_16x16x32_f16(
          false, a1.v, false, b[g].v, (short)0, acc[1][g].v, false, false);
    }
  }

  if (!ROPE) {
#pragma unroll
    for (int mi = 0; mi < 2; ++mi) {
      const int rbase = m0 + mi * 16 + koff;
#pragma unroll
      for (int g = 0; g < 4; ++g) {
        const int col = n0 + g * 16 + l15;
        const float bv = bias[col];
#pragma unroll
        for (int r = 0; r < 8; ++r)
          C[(size_t)(rbase + r) * N + col] = acc[mi][g].f[r] + bv;
      }
    }
    return;
  }

  // --- fused YaRN RoPE + q/k/v split (wave's 64-col span is head-aligned) ---
#pragma unroll
  for (int mi = 0; mi < 2; ++mi) {
    const int rbase = m0 + mi * 16 + koff;
    if (n0 < Q_DIM) {
      const int h = n0 >> 6;
#pragma unroll
      for (int g = 0; g < 2; ++g) {
        const int d  = g * 16 + l15;                 // 0..31 (pair index)
        const float b1 = bias[n0 + g * 16 + l15];
        const float b2 = bias[n0 + (g + 2) * 16 + l15];
#pragma unroll
        for (int r = 0; r < 8; ++r) {
          const int row = rbase + r;
          float c, sn;
          yarn_cs(d, (float)row, c, sn);
          const float x1 = acc[mi][g].f[r] + b1;
          const float x2 = acc[mi][g + 2].f[r] + b2;
          _Float16* dst = qf + ((size_t)h * SEQ + row) * HEAD_DIM;
          dst[d]      = (_Float16)(x1 * c - x2 * sn);
          dst[d + 32] = (_Float16)(x2 * c + x1 * sn);
        }
      }
    } else if (n0 < Q_DIM + KV_DIM) {
      const int h = (n0 - Q_DIM) >> 6;
#pragma unroll
      for (int g = 0; g < 2; ++g) {
        const int d  = g * 16 + l15;
        const float b1 = bias[n0 + g * 16 + l15];
        const float b2 = bias[n0 + (g + 2) * 16 + l15];
#pragma unroll
        for (int r = 0; r < 8; ++r) {
          const int row = rbase + r;
          float c, sn;
          yarn_cs(d, (float)row, c, sn);
          const float x1 = acc[mi][g].f[r] + b1;
          const float x2 = acc[mi][g + 2].f[r] + b2;
          _Float16* dst = kf + ((size_t)h * SEQ + row) * HEAD_DIM;
          dst[d]      = (_Float16)(x1 * c - x2 * sn);
          dst[d + 32] = (_Float16)(x2 * c + x1 * sn);
        }
      }
    } else {
      const int h = (n0 - Q_DIM - KV_DIM) >> 6;
#pragma unroll
      for (int g = 0; g < 4; ++g) {
        const int d = g * 16 + l15;
        const float bv = bias[n0 + g * 16 + l15];
#pragma unroll
        for (int r = 0; r < 8; ++r) {
          const int row = rbase + r;
          vT[((size_t)h * HEAD_DIM + d) * SEQ + row] =
              (_Float16)(acc[mi][g].f[r] + bv);
        }
      }
    }
  }
}

// ---------------------------------------------------------------------------
// Async staging of one 16-key K tile (16x64) and V tile (64x16) into LDS.
// 8 x b128 per lane-set = 8 async ops per call.
// ---------------------------------------------------------------------------
#if defined(USE_ASYNC_LDS)
__device__ __forceinline__ void stage_kv(const _Float16* kf, const _Float16* vT,
                                         int kvh, int kb, int lane,
                                         _Float16* kd, _Float16* vd) {
#pragma unroll
  for (int i = 0; i < 4; ++i) {
    const int key = i * 8 + (lane >> 2);
    const int off = (lane & 3) * 16;
    const _Float16* src = kf + ((size_t)kvh * SEQ + kb + key) * HEAD_DIM + off;
    async_ld16(src, kd + key * HEAD_DIM + off);
  }
#pragma unroll
  for (int i = 0; i < 4; ++i) {
    const int d   = i * 16 + (lane >> 1);
    const int off = (lane & 1) * 8;
    const _Float16* src = vT + ((size_t)kvh * HEAD_DIM + d) * SEQ + kb + off;
    async_ld16(src, vd + d * 16 + off);
  }
}
#endif

// ---------------------------------------------------------------------------
// Sliding-window causal GQA with sink logit.  One wave per (head, 16-query
// block).  QK^T: 2 chained WMMAs (K=64); online softmax with in-half shuffle
// reductions; P transposed through LDS; PV: 4 WMMAs over head_dim=64.
// K/V tiles double-buffered in LDS via async-to-LDS when available.
// ---------------------------------------------------------------------------
__global__ void __launch_bounds__(32)
attn_swa(const _Float16* __restrict__ qf, const _Float16* __restrict__ kf,
         const _Float16* __restrict__ vT, const float* __restrict__ sinks,
         _Float16* __restrict__ attn) {
#if defined(USE_ASYNC_LDS)
  __shared__ __align__(16) _Float16 kbuf[2][16 * HEAD_DIM];
  __shared__ __align__(16) _Float16 vbuf[2][HEAD_DIM * 16];
#endif
  __shared__ __align__(16) _Float16 plds[16 * 16];

  const int lane = threadIdx.x;
  const int q0   = blockIdx.x * 16;
  const int h    = blockIdx.y;
  const int kvh  = h >> 3;
  const int half = lane >> 4;
  const int l15  = lane & 15;
  const int koff = half * 8;

  FragAB qa0, qa1;
  {
    const _Float16* qb = qf + ((size_t)h * SEQ + (q0 + l15)) * HEAD_DIM;
    qa0.h2[0] = *(const v8h*)(qb + koff);
    qa0.h2[1] = *(const v8h*)(qb + koff + 16);
    qa1.h2[0] = *(const v8h*)(qb + koff + 32);
    qa1.h2[1] = *(const v8h*)(qb + koff + 48);
  }

  FragC acc[4];
#pragma unroll
  for (int g = 0; g < 4; ++g) acc[g].v = (v8f){};
  float mrow[8], lrow[8];
#pragma unroll
  for (int r = 0; r < 8; ++r) { mrow[r] = -1e30f; lrow[r] = 0.f; }

  int kb0 = q0 - WINDOW;
  if (kb0 < 0) kb0 = 0;
  const v8h zero8 = {};

#if defined(USE_ASYNC_LDS)
  stage_kv(kf, vT, kvh, kb0, lane, kbuf[0], vbuf[0]);
  int buf = 0;
#endif

  for (int kb = kb0; kb <= q0; kb += 16) {
    const bool more = (kb + 16 <= q0);
    FragAB kf0, kf1;
#if defined(USE_ASYNC_LDS)
    if (more) {
      wait_dscnt0();                              // WAR: prior reads of buf^1
      stage_kv(kf, vT, kvh, kb + 16, lane, kbuf[buf ^ 1], vbuf[buf ^ 1]);
      wait_asynccnt<8>();                         // current buffer complete
    } else {
      wait_asynccnt<0>();
    }
    const _Float16* kp = &kbuf[buf][l15 * HEAD_DIM + koff];
    kf0.h2[0] = *(const v8h*)(kp);
    kf0.h2[1] = *(const v8h*)(kp + 16);
    kf1.h2[0] = *(const v8h*)(kp + 32);
    kf1.h2[1] = *(const v8h*)(kp + 48);
#else
    const _Float16* kp = kf + ((size_t)kvh * SEQ + (kb + l15)) * HEAD_DIM;
    kf0.h2[0] = *(const v8h*)(kp + koff);
    kf0.h2[1] = *(const v8h*)(kp + koff + 16);
    kf1.h2[0] = *(const v8h*)(kp + koff + 32);
    kf1.h2[1] = *(const v8h*)(kp + koff + 48);
#endif

    v8f sc = {};
    sc = __builtin_amdgcn_wmma_f32_16x16x32_f16(false, qa0.v, false, kf0.v,
                                                (short)0, sc, false, false);
    sc = __builtin_amdgcn_wmma_f32_16x16x32_f16(false, qa1.v, false, kf1.v,
                                                (short)0, sc, false, false);
    FragC s8; s8.v = sc;

    const int key = kb + l15;
    float pvals[8], sfac[8];
#pragma unroll
    for (int r = 0; r < 8; ++r) {
      const int q = q0 + half * 8 + r;
      const bool ok = (key <= q) && ((q - key) < WINDOW);
      const float sval = s8.f[r] * SM_SCALE;
      float sm = ok ? sval : -1e30f;
      sm = fmaxf(sm, __shfl_xor(sm, 1, 32));
      sm = fmaxf(sm, __shfl_xor(sm, 2, 32));
      sm = fmaxf(sm, __shfl_xor(sm, 4, 32));
      sm = fmaxf(sm, __shfl_xor(sm, 8, 32));
      const float mn = fmaxf(mrow[r], sm);
      const float sf = __expf(mrow[r] - mn);
      const float p  = ok ? __expf(sval - mn) : 0.f;
      float rs = p;
      rs += __shfl_xor(rs, 1, 32);
      rs += __shfl_xor(rs, 2, 32);
      rs += __shfl_xor(rs, 4, 32);
      rs += __shfl_xor(rs, 8, 32);
      lrow[r] = lrow[r] * sf + rs;
      mrow[r] = mn;
      pvals[r] = p;
      sfac[r]  = sf;
    }
#pragma unroll
    for (int g = 0; g < 4; ++g)
#pragma unroll
      for (int r = 0; r < 8; ++r) acc[g].f[r] *= sfac[r];

#pragma unroll
    for (int r = 0; r < 8; ++r)
      plds[(half * 8 + r) * 16 + l15] = (_Float16)pvals[r];
    __syncthreads();

    FragAB pa;
    pa.h2[0] = *(const v8h*)&plds[l15 * 16 + koff];
    pa.h2[1] = zero8;

#pragma unroll
    for (int g = 0; g < 4; ++g) {
      FragAB vb;
#if defined(USE_ASYNC_LDS)
      vb.h2[0] = *(const v8h*)&vbuf[buf][(g * 16 + l15) * 16 + koff];
#else
      const _Float16* vp =
          vT + ((size_t)kvh * HEAD_DIM + (g * 16 + l15)) * SEQ + kb + koff;
      vb.h2[0] = *(const v8h*)vp;
#endif
      vb.h2[1] = zero8;
      acc[g].v = __builtin_amdgcn_wmma_f32_16x16x32_f16(
          false, pa.v, false, vb.v, (short)0, acc[g].v, false, false);
    }
    __syncthreads();
#if defined(USE_ASYNC_LDS)
    buf ^= 1;
#endif
  }

  const float snk = sinks[h];
#pragma unroll
  for (int r = 0; r < 8; ++r) {
    const float mn    = fmaxf(mrow[r], snk);
    const float corr  = __expf(mrow[r] - mn);
    const float denom = lrow[r] * corr + __expf(snk - mn);
    const float scale = corr / denom;
    const int row = q0 + half * 8 + r;
#pragma unroll
    for (int g = 0; g < 4; ++g)
      attn[(size_t)row * Q_DIM + h * HEAD_DIM + g * 16 + l15] =
          (_Float16)(acc[g].f[r] * scale);
  }
}

// ---------------------------------------------------------------------------
// Launch sequence.  Workspace (256B-aligned sections, ~90 MB):
//   t_f16 | qkvw_f16 | outw_f16 | q_f16 | k_f16 | vT_f16 | attn_f16
// ---------------------------------------------------------------------------
extern "C" void kernel_launch(void* const* d_in, const int* in_sizes, int n_in,
                              void* d_out, int out_size, void* d_ws, size_t ws_size,
                              hipStream_t stream) {
  (void)in_sizes; (void)n_in; (void)out_size; (void)ws_size;
  const float* x      = (const float*)d_in[0];
  const float* sinks  = (const float*)d_in[1];
  const float* norm_w = (const float*)d_in[2];
  const float* qkv_w  = (const float*)d_in[3];
  const float* qkv_b  = (const float*)d_in[4];
  const float* out_w  = (const float*)d_in[5];
  const float* out_b  = (const float*)d_in[6];
  float* out = (float*)d_out;

  char* ws = (char*)d_ws;
  size_t off = 0;
  _Float16* t16  = (_Float16*)(ws + off); off += (size_t)SEQ * DIM * 2;
  _Float16* qw16 = (_Float16*)(ws + off); off += (size_t)QKV_DIM * DIM * 2;
  _Float16* ow16 = (_Float16*)(ws + off); off += (size_t)DIM * Q_DIM * 2;
  _Float16* q16  = (_Float16*)(ws + off); off += (size_t)N_HEADS * SEQ * HEAD_DIM * 2;
  _Float16* k16  = (_Float16*)(ws + off); off += (size_t)N_KV * SEQ * HEAD_DIM * 2;
  _Float16* vT16 = (_Float16*)(ws + off); off += (size_t)N_KV * HEAD_DIM * SEQ * 2;
  _Float16* at16 = (_Float16*)(ws + off);

  // 1. RMSNorm + cast
  rmsnorm_cast<<<SEQ, 256, 0, stream>>>(x, norm_w, t16);

  // 2. weight casts (f16 copies stay resident in 192MB L2)
  {
    int n1 = QKV_DIM * DIM;
    cast_f16<<<(n1 + 255) / 256, 256, 0, stream>>>(qkv_w, qw16, n1);
    int n2 = DIM * Q_DIM;
    cast_f16<<<(n2 + 255) / 256, 256, 0, stream>>>(out_w, ow16, n2);
  }

  // 3. QKV projection with fused YaRN RoPE + split + V transpose
  {
    dim3 grid(QKV_DIM / 64, SEQ / 256);
    gemm_nt_f16<true><<<grid, 256, 0, stream>>>(t16, qw16, qkv_b, nullptr,
                                                q16, k16, vT16,
                                                SEQ, QKV_DIM, DIM);
  }

  // 4. sliding-window GQA with sinks
  attn_swa<<<dim3(SEQ / 16, N_HEADS), 32, 0, stream>>>(q16, k16, vT16, sinks, at16);

  // 5. output projection -> d_out
  {
    dim3 grid(DIM / 64, SEQ / 256);
    gemm_nt_f16<false><<<grid, 256, 0, stream>>>(at16, ow16, out_b, out,
                                                 nullptr, nullptr, nullptr,
                                                 SEQ, DIM, Q_DIM);
  }
}
